// CRG_30545807409555
// MI455X (gfx1250) — matrix operand order repaired and checked by reading, not compiled, not run on hardware
//
#include <hip/hip_runtime.h>

typedef __attribute__((ext_vector_type(16))) _Float16 v16h;
typedef __attribute__((ext_vector_type(8)))  _Float16 v8h;
typedef __attribute__((ext_vector_type(8)))  float    v8f;

#define NEGBIG (-1e9f)

// ---------------------------------------------------------------------------
// Stage 1a: h1 = relu(coords @ W_h1 + b_h1)   [BN,3] @ [3,64] -> [BN,64]
// ---------------------------------------------------------------------------
__global__ void k_h1(const float* __restrict__ coords, const float* __restrict__ W,
                     const float* __restrict__ b, float* __restrict__ out, int BN) {
  int t = blockIdx.x * blockDim.x + threadIdx.x;
  if (t >= BN * 64) return;
  int node = t >> 6, o = t & 63;
  float c0 = coords[node * 3 + 0], c1 = coords[node * 3 + 1], c2 = coords[node * 3 + 2];
  float s = c0 * W[0 * 64 + o] + c1 * W[1 * 64 + o] + c2 * W[2 * 64 + o] + b[o];
  out[t] = fmaxf(s, 0.f);
}

// ---------------------------------------------------------------------------
// Stage 1b: X[:, :64] = relu(h1 @ W_h2 + b_h2); X[:,64:127]=feat; X[:,127]=score
// ---------------------------------------------------------------------------
__global__ void k_h2cat(const float* __restrict__ h1, const float* __restrict__ W,
                        const float* __restrict__ b, const float* __restrict__ feat,
                        const float* __restrict__ score, float* __restrict__ X, int BN) {
  int t = blockIdx.x * blockDim.x + threadIdx.x;
  if (t >= BN * 128) return;
  int node = t >> 7, o = t & 127;
  float r;
  if (o < 64) {
    const float* h = h1 + (size_t)node * 64;
    float s = b[o];
    for (int f = 0; f < 64; ++f) s += h[f] * W[f * 64 + o];
    r = fmaxf(s, 0.f);
  } else if (o < 127) {
    r = feat[(size_t)node * 63 + (o - 64)];
  } else {
    r = score[node];
  }
  X[t] = r;
}

// ---------------------------------------------------------------------------
// EdgeConv factorization (f32 accumulate, f16 store):
//   u_i = x_i @ (W1[:F]-W1[F:]) + b1 ; v_j = x_j @ W1[F:]
// ---------------------------------------------------------------------------
__global__ void k_uv(const float* __restrict__ X, const float* __restrict__ W1,
                     const float* __restrict__ b1, _Float16* __restrict__ U,
                     _Float16* __restrict__ V, int BN, int F) {
  int t = blockIdx.x * blockDim.x + threadIdx.x;
  if (t >= BN * 64) return;
  int node = t >> 6, o = t & 63;
  const float* x = X + (size_t)node * F;
  float su = 0.f, sv = 0.f;
  for (int f = 0; f < F; ++f) {
    float xv = x[f];
    float wa = W1[f * 64 + o];
    float wb = W1[(F + f) * 64 + o];
    su += xv * (wa - wb);
    sv += xv * wb;
  }
  U[t] = (_Float16)(su + b1[o]);
  V[t] = (_Float16)sv;
}

// Load a v16h A/K fragment slice from a 64-half row: two contiguous 8-half
// runs selected by lane half (ISA 16-bit A layout).
__device__ __forceinline__ v16h load_frag_row(const _Float16* __restrict__ row,
                                              int kt, int laneHi) {
  v8h lo = *(const v8h*)(row + kt * 32 + laneHi * 8);
  v8h hi = *(const v8h*)(row + kt * 32 + 16 + laneHi * 8);
  return __builtin_shufflevector(lo, hi, 0, 1, 2, 3, 4, 5, 6, 7,
                                         8, 9, 10, 11, 12, 13, 14, 15);
}

// ---------------------------------------------------------------------------
// EdgeConv core: Y[i,:] = max_{j: adj[i,j]>0} relu( relu(u_i + v_j) @ W2 + b2 )
// One wave32 per node i; 4 waves per block. Dense over j in 16-row tiles via
// v_wmma_f32_16x16x32_f16. A built with packed f16 VOP3P ops
// (v_pk_add_f16 / v_pk_max_num_f16); W2 kept resident as 8 B-fragments.
// ---------------------------------------------------------------------------
__global__ __launch_bounds__(128) void k_edge(
    const _Float16* __restrict__ U, const _Float16* __restrict__ V,
    const float* __restrict__ adj, const float* __restrict__ W2,
    const float* __restrict__ b2, float* __restrict__ Y, int Nc) {
  const int lane   = threadIdx.x & 31;
  const int wave   = threadIdx.x >> 5;
  const int i      = blockIdx.x * 4 + wave;   // global node id in [0, B*N)
  const int bb     = i / Nc;
  const int nn     = i % Nc;
  const int laneN  = lane & 15;
  const int laneHi = lane >> 4;

  // Preload W2 as 2(K-tiles) x 4(N-tiles) f16 B-fragments (ISA B layout:
  // elements 0..7 -> K laneHi*8+e, elements 8..15 -> K 16+laneHi*8+(e-8)).
  v16h Bf[2][4];
#pragma unroll
  for (int kt = 0; kt < 2; ++kt)
#pragma unroll
    for (int nt = 0; nt < 4; ++nt)
#pragma unroll
      for (int e = 0; e < 16; ++e) {
        int k = kt * 32 + (e < 8 ? laneHi * 8 + e : 16 + laneHi * 8 + (e - 8));
        Bf[kt][nt][e] = (_Float16)W2[k * 64 + nt * 16 + laneN];
      }

  // Cache u_i pre-swizzled into the A-fragment lane layout (f16)
  const _Float16* Ui = U + (size_t)i * 64;
  v16h uf[2];
#pragma unroll
  for (int kt = 0; kt < 2; ++kt) uf[kt] = load_frag_row(Ui, kt, laneHi);

  float bias[4];
#pragma unroll
  for (int nt = 0; nt < 4; ++nt) bias[nt] = b2[nt * 16 + laneN];

  float acc[4] = {NEGBIG, NEGBIG, NEGBIG, NEGBIG};
  const v16h zero16 = {};

  const _Float16* Vb  = V + (size_t)bb * Nc * 64;
  const float* adjRow = adj + (size_t)bb * Nc * Nc + (size_t)nn * Nc;

  const int nJT = Nc / 16;
  for (int jt = 0; jt < nJT; ++jt) {
    int jrow = jt * 16 + laneN;
    const _Float16* Vj = Vb + (size_t)jrow * 64;
    if (jt + 1 < nJT)  // global_prefetch_b8 of next tile's v rows
      __builtin_prefetch(Vb + (size_t)(jrow + 16) * 64, 0, 1);

    // A fragments: relu(u_i + v_j) via packed f16 ops
    v16h A[2];
#pragma unroll
    for (int kt = 0; kt < 2; ++kt) {
      v16h s = uf[kt] + load_frag_row(Vj, kt, laneHi);  // v_pk_add_f16
      A[kt] = __builtin_elementwise_max(s, zero16);     // v_pk_max_num_f16
    }

    // Adjacency values for this lane's C rows (row M = laneHi*8 + r)
    float am[8];
#pragma unroll
    for (int r = 0; r < 8; ++r) am[r] = adjRow[jt * 16 + laneHi * 8 + r];

#pragma unroll
    for (int nt = 0; nt < 4; ++nt) {
      v8f c = {};
      c = __builtin_amdgcn_wmma_f32_16x16x32_f16(false, A[0], false, Bf[0][nt],
                                                 (short)0, c, false, false);
      c = __builtin_amdgcn_wmma_f32_16x16x32_f16(false, A[1], false, Bf[1][nt],
                                                 (short)0, c, false, false);
      // masked max over the 8 rows this lane holds (f32)
      float m = NEGBIG;
#pragma unroll
      for (int r = 0; r < 8; ++r) {
        float h = fmaxf(c[r] + bias[nt], 0.f);
        m = (am[r] > 0.f) ? fmaxf(m, h) : m;
      }
      // combine row halves: lanes L and L+16 hold same column
      m = fmaxf(m, __shfl_xor(m, 16, 32));
      acc[nt] = fmaxf(acc[nt], m);
    }
  }

  if (lane < 16) {
    float* Yi = Y + (size_t)i * 64;
#pragma unroll
    for (int nt = 0; nt < 4; ++nt) Yi[nt * 16 + laneN] = acc[nt];
  }
}

// ---------------------------------------------------------------------------
// Elementwise residual add: out = a + b
// ---------------------------------------------------------------------------
__global__ void k_add(const float* __restrict__ a, const float* __restrict__ b,
                      float* __restrict__ out, int n) {
  int t = blockIdx.x * blockDim.x + threadIdx.x;
  if (t < n) out[t] = a[t] + b[t];
}

// ---------------------------------------------------------------------------
// Head: H = relu(Z @ W_h6 + b_h6)   [BN,64] @ [64,128]
// ---------------------------------------------------------------------------
__global__ void k_h6(const float* __restrict__ Z, const float* __restrict__ W,
                     const float* __restrict__ b, float* __restrict__ H, int BN) {
  int t = blockIdx.x * blockDim.x + threadIdx.x;
  if (t >= BN * 128) return;
  int node = t >> 7, o = t & 127;
  const float* z = Z + (size_t)node * 64;
  float s = b[o];
  for (int f = 0; f < 64; ++f) s += z[f] * W[f * 128 + o];
  H[t] = fmaxf(s, 0.f);
}

// ---------------------------------------------------------------------------
// Head: out = relu(H @ W_out + b_out)   [BN,128] @ [128,3]
// ---------------------------------------------------------------------------
__global__ void k_out(const float* __restrict__ H, const float* __restrict__ W,
                      const float* __restrict__ b, float* __restrict__ out, int BN) {
  int t = blockIdx.x * blockDim.x + threadIdx.x;
  if (t >= BN * 3) return;
  int node = t / 3, o = t % 3;
  const float* h = H + (size_t)node * 128;
  float s = b[o];
  for (int f = 0; f < 128; ++f) s += h[f] * W[f * 3 + o];
  out[t] = fmaxf(s, 0.f);
}

// ---------------------------------------------------------------------------
extern "C" void kernel_launch(void* const* d_in, const int* in_sizes, int n_in,
                              void* d_out, int out_size, void* d_ws, size_t ws_size,
                              hipStream_t stream) {
  const float* coords = (const float*)d_in[0];
  const float* adj    = (const float*)d_in[1];
  const float* feat   = (const float*)d_in[2];
  const float* score  = (const float*)d_in[3];
  const float* W_h1   = (const float*)d_in[4];
  const float* b_h1   = (const float*)d_in[5];
  const float* W_h2   = (const float*)d_in[6];
  const float* b_h2   = (const float*)d_in[7];
  const float* ecW1[3] = {(const float*)d_in[8],  (const float*)d_in[12], (const float*)d_in[16]};
  const float* ecb1[3] = {(const float*)d_in[9],  (const float*)d_in[13], (const float*)d_in[17]};
  const float* ecW2[3] = {(const float*)d_in[10], (const float*)d_in[14], (const float*)d_in[18]};
  const float* ecb2[3] = {(const float*)d_in[11], (const float*)d_in[15], (const float*)d_in[19]};
  const float* W_h6  = (const float*)d_in[20];
  const float* b_h6  = (const float*)d_in[21];
  const float* W_out = (const float*)d_in[22];
  const float* b_out = (const float*)d_in[23];

  const int BN = in_sizes[0] / 3;        // B*N = 1024
  const int N  = in_sizes[1] / BN;       // 512

  // Workspace layout (float granularity): ~1.75 MB total
  float*     w  = (float*)d_ws;
  float*     A0 = w;                                  // [BN,128] X / head hidden
  _Float16*  Uh = (_Float16*)(w + (size_t)BN * 128);  // [BN,64] f16
  _Float16*  Vh = (_Float16*)(w + (size_t)BN * 192);  // [BN,64] f16
  float*     Y0 = w + (size_t)BN * 256;               // [BN,64] ec1_out
  float*     Y1 = w + (size_t)BN * 320;               // [BN,64] ec2_out
  float*     Y2 = w + (size_t)BN * 384;               // [BN,64] working / h1 temp

  const int T = 256;
  // Node MLP + concat -> X (A0); Y2 region is free here, use as h1 temp
  k_h1<<<(BN * 64 + T - 1) / T, T, 0, stream>>>(coords, W_h1, b_h1, Y2, BN);
  k_h2cat<<<(BN * 128 + T - 1) / T, T, 0, stream>>>(Y2, W_h2, b_h2, feat, score, A0, BN);

  // EdgeConv1 (F=128)
  k_uv<<<(BN * 64 + T - 1) / T, T, 0, stream>>>(A0, ecW1[0], ecb1[0], Uh, Vh, BN, 128);
  k_edge<<<BN / 4, 128, 0, stream>>>(Uh, Vh, adj, ecW2[0], ecb2[0], Y0, N);

  // EdgeConv2 (F=64)
  k_uv<<<(BN * 64 + T - 1) / T, T, 0, stream>>>(Y0, ecW1[1], ecb1[1], Uh, Vh, BN, 64);
  k_edge<<<BN / 4, 128, 0, stream>>>(Uh, Vh, adj, ecW2[1], ecb2[1], Y1, N);

  // residual: Y2 = Y1 + Y0
  k_add<<<(BN * 64 + T - 1) / T, T, 0, stream>>>(Y1, Y0, Y2, BN * 64);

  // EdgeConv3 (F=64) — output overwrites Y2 after u/v extracted
  k_uv<<<(BN * 64 + T - 1) / T, T, 0, stream>>>(Y2, ecW1[2], ecb1[2], Uh, Vh, BN, 64);
  k_edge<<<BN / 4, 128, 0, stream>>>(Uh, Vh, adj, ecW2[2], ecb2[2], Y2, N);

  // residual: Y2 = Y2 + Y1
  k_add<<<(BN * 64 + T - 1) / T, T, 0, stream>>>(Y2, Y1, Y2, BN * 64);

  // Head MLPs
  k_h6<<<(BN * 128 + T - 1) / T, T, 0, stream>>>(Y2, W_h6, b_h6, A0, BN);
  k_out<<<(BN * 3 + T - 1) / T, T, 0, stream>>>(A0, W_out, b_out, (float*)d_out, BN);
}
